// RecursiveNN_52089363365970
// MI455X (gfx1250) — matrix operand order repaired and checked by reading, not compiled, not run on hardware
//
#include <hip/hip_runtime.h>
#include <hip/hip_bf16.h>

#define HID 128
#define IN_K 64
#define N_LEAVES 262144

typedef __attribute__((ext_vector_type(16))) _Float16 v16h;
typedef __attribute__((ext_vector_type(8)))  _Float16 v8h;
typedef __attribute__((ext_vector_type(8)))  float    v8f;

union AF {
    v16h v16;
    v8h  v8[2];
    _Float16 h[16];
};

// Branch-free tanh: tanh(x) = 2/(1+e^{-2x}) - 1, via v_exp_f32 + v_rcp_f32.
// Limits are exact: x->+inf => t->0 => 1 ; x->-inf => t->inf => -1. No clamps,
// no EXEC-mask branching (keeps the epilogue pure VALU).
static __device__ __forceinline__ float fast_tanh(float x) {
    const float t = __builtin_amdgcn_exp2f(x * -2.885390081777927f); // e^{-2x}
    return __builtin_fmaf(2.0f, __builtin_amdgcn_rcpf(1.0f + t), -1.0f);
}

// ---------------------------------------------------------------------------
// Tensor Data Mover staging: 1-D tile of f16 elements, global -> LDS.
// D# layout per CDNA5 ISA (group0/group1; groups 2/3 zero
// since tile_dim2 == 0). Issued by one wave; waiter uses s_wait_tensorcnt.
// This toolchain exposes the 6-arg builtin: (v4u, v8i, v4i, v4i, v8i, cpol).
// ---------------------------------------------------------------------------
#if __has_builtin(__builtin_amdgcn_tensor_load_to_lds) && __has_builtin(__builtin_amdgcn_s_wait_tensorcnt)
#define USE_TDM 1
typedef unsigned int v4u __attribute__((ext_vector_type(4)));
typedef int          v8i __attribute__((ext_vector_type(8)));
typedef int          v4i __attribute__((ext_vector_type(4)));

static __device__ __forceinline__ void tdm_load_f16_1d(void* lds_dst, const void* gsrc,
                                                       unsigned int nelem) {
    const unsigned long long ga = (unsigned long long)(uintptr_t)gsrc;
    const unsigned int la = (unsigned int)(uintptr_t)lds_dst; // flat LDS addr[31:0] == LDS offset
    v4u g0;
    g0[0] = 1u;                                               // count=1 (valid user D#)
    g0[1] = la;                                               // lds_addr
    g0[2] = (unsigned int)(ga & 0xffffffffu);                 // global_addr[31:0]
    g0[3] = (unsigned int)((ga >> 32) & 0x1ffffffu) | (2u << 30); // global_addr[56:32] | type=2
    unsigned int w[8];
    w[0] = 1u << 16;                                          // wg_mask=0, data_size=1 (2B)
    w[1] = (nelem & 0xffffu) << 16;                           // tensor_dim0[15:0] @ bits 63:48
    w[2] = ((nelem >> 16) & 0xffffu) | (1u << 16);            // tensor_dim0[31:16], tensor_dim1=1
    w[3] = (nelem & 0xffffu) << 16;                           // tile_dim0 = nelem @ bits 127:112
    w[4] = 1u;                                                // tile_dim1=1, tile_dim2=0
    w[5] = nelem;                                             // tensor_dim0_stride[31:0]
    w[6] = (nelem & 0xffffu) << 16;                           // stride0 hi=0, stride1[15:0]
    w[7] = nelem >> 16;                                       // stride1[47:16]
    v8i g1;
#pragma unroll
    for (int i = 0; i < 8; ++i) g1[i] = (int)w[i];
    const v4i z4 = {0, 0, 0, 0};
    const v8i z8 = {0, 0, 0, 0, 0, 0, 0, 0};
    __builtin_amdgcn_tensor_load_to_lds(g0, g1, z4, z4, z8, 0);
}
#endif

// Load an A-matrix fragment (16x32 f16, contiguous k order, same as the B loads) for this lane.
// Lane half hs holds K {k0..k0+15} with k0 = ks*32 + hs*16.
static __device__ __forceinline__ v16h load_a_frag(const _Float16* __restrict__ p, int k0) {
    AF a;
    a.v8[0] = *(const v8h*)(p + k0);
    a.v8[1] = *(const v8h*)(p + k0 + 8);
    return a.v16;
}

// One 16-row output block of an internal tree level:
// out[row] = tanh(in[2*row] @ W_l + in[2*row+1] @ W_r + bias)
static __device__ __forceinline__ void block16(const _Float16* __restrict__ in,
                                               _Float16* __restrict__ out,
                                               const _Float16* __restrict__ WlT,
                                               const _Float16* __restrict__ WrT,
                                               const float* __restrict__ bias,
                                               int rowBase) {
    const int lane = threadIdx.x & 31;
    const int hs   = lane >> 4;
    const int m    = lane & 15;
    const int row  = rowBase + m;

    const _Float16* rl = in + (size_t)(2 * row) * HID;
    const _Float16* rr = rl + HID;

    v16h aL[4], aR[4];
#pragma unroll
    for (int ks = 0; ks < 4; ++ks) {
        const int k0 = ks * 32 + hs * 16;
        aL[ks] = load_a_frag(rl, k0);
        aR[ks] = load_a_frag(rr, k0);
    }

#pragma unroll
    for (int nt = 0; nt < 8; ++nt) {
        v8f c = {};
#pragma unroll
        for (int ks = 0; ks < 4; ++ks) {
            const int boff = (nt * 16 + m) * HID + ks * 32 + hs * 16;
            v16h bL = *(const v16h*)(WlT + boff);
            v16h bR = *(const v16h*)(WrT + boff);
            c = __builtin_amdgcn_wmma_f32_16x16x32_f16(false, aL[ks], false, bL,
                                                       (short)0, c, false, false);
            c = __builtin_amdgcn_wmma_f32_16x16x32_f16(false, aR[ks], false, bR,
                                                       (short)0, c, false, false);
        }
        const float bv = bias[nt * 16 + m];
#pragma unroll
        for (int r = 0; r < 8; ++r) {
            const int M = r + hs * 8;
            out[(size_t)(rowBase + M) * HID + nt * 16 + m] = (_Float16)fast_tanh(c[r] + bv);
        }
    }
}

// Leaf block: out[row] = tanh(x[row] @ W_in + b_in), fp32 x converted on the fly.
static __device__ __forceinline__ void leaf_block16(const float* __restrict__ x,
                                                    _Float16* __restrict__ out,
                                                    const _Float16* __restrict__ WinT,
                                                    const float* __restrict__ bin,
                                                    int rowBase) {
    const int lane = threadIdx.x & 31;
    const int hs   = lane >> 4;
    const int m    = lane & 15;
    const float* xr = x + (size_t)(rowBase + m) * IN_K;

    v16h a[2];
#pragma unroll
    for (int ks = 0; ks < 2; ++ks) {
        const int k0 = ks * 32 + hs * 16;
        AF af;
#pragma unroll
        for (int j = 0; j < 8; ++j) {
            af.h[j]     = (_Float16)xr[k0 + j];
            af.h[8 + j] = (_Float16)xr[k0 + 8 + j];
        }
        a[ks] = af.v16;
    }

#pragma unroll
    for (int nt = 0; nt < 8; ++nt) {
        v8f c = {};
#pragma unroll
        for (int ks = 0; ks < 2; ++ks) {
            const int boff = (nt * 16 + m) * IN_K + ks * 32 + hs * 16;
            v16h b = *(const v16h*)(WinT + boff);
            c = __builtin_amdgcn_wmma_f32_16x16x32_f16(false, a[ks], false, b,
                                                       (short)0, c, false, false);
        }
        const float bv = bin[nt * 16 + m];
#pragma unroll
        for (int r = 0; r < 8; ++r) {
            const int M = r + hs * 8;
            out[(size_t)(rowBase + M) * HID + nt * 16 + m] = (_Float16)fast_tanh(c[r] + bv);
        }
    }
}

// ---------------------------------------------------------------------------
// Kernels
// ---------------------------------------------------------------------------

__global__ void RecNN_prep_kernel(const float* __restrict__ Win, const float* __restrict__ Wl,
                                  const float* __restrict__ Wr, const float* __restrict__ bl,
                                  const float* __restrict__ br,
                                  _Float16* __restrict__ WinT, _Float16* __restrict__ WlT,
                                  _Float16* __restrict__ WrT, float* __restrict__ biasLR) {
    const int idx = blockIdx.x * blockDim.x + threadIdx.x;
    if (idx < HID * HID) {
        const int n = idx / HID, k = idx % HID;
        WlT[n * HID + k] = (_Float16)Wl[k * HID + n];
        WrT[n * HID + k] = (_Float16)Wr[k * HID + n];
    }
    if (idx < HID * IN_K) {
        const int n = idx / IN_K, k = idx % IN_K;
        WinT[n * IN_K + k] = (_Float16)Win[k * HID + n];
    }
    if (idx < HID) biasLR[idx] = bl[idx] + br[idx];
}

__global__ void RecNN_leaf_kernel(const float* __restrict__ x, _Float16* __restrict__ out,
                                  const _Float16* __restrict__ WinT,
                                  const float* __restrict__ bin) {
    extern __shared__ char smem[];
    _Float16* w = (_Float16*)smem;                       // 16 KB: W_in^T
#if defined(USE_TDM)
    if (threadIdx.x < 32) {                              // wave 0 issues the TDM
        tdm_load_f16_1d(w, WinT, HID * IN_K);
        __builtin_amdgcn_s_wait_tensorcnt(0);
    }
#else
    {
        const uint4* s = (const uint4*)WinT;
        uint4* d = (uint4*)w;
        for (int i = threadIdx.x; i < (HID * IN_K) / 8; i += blockDim.x) d[i] = s[i];
    }
#endif
    __syncthreads();
    const int wave = threadIdx.x >> 5;
    const int rowBase = (blockIdx.x * 8 + wave) * 16;
    leaf_block16(x, out, w, bin, rowBase);
}

__global__ void RecNN_level_kernel(const _Float16* __restrict__ in, _Float16* __restrict__ out,
                                   const _Float16* __restrict__ WlT,
                                   const _Float16* __restrict__ WrT,
                                   const float* __restrict__ bias, int Mout) {
    extern __shared__ char smem[];
    _Float16* wl = (_Float16*)smem;                      // 32 KB
    _Float16* wr = wl + HID * HID;                       // 32 KB
#if defined(USE_TDM)
    if (threadIdx.x < 32) {                              // wave 0 issues both TDMs
        tdm_load_f16_1d(wl, WlT, HID * HID);
        tdm_load_f16_1d(wr, WrT, HID * HID);
        __builtin_amdgcn_s_wait_tensorcnt(0);
    }
#else
    {
        const uint4* s1 = (const uint4*)WlT;
        const uint4* s2 = (const uint4*)WrT;
        uint4* d1 = (uint4*)wl;
        uint4* d2 = (uint4*)wr;
        for (int i = threadIdx.x; i < (HID * HID) / 8; i += blockDim.x) {
            d1[i] = s1[i];
            d2[i] = s2[i];
        }
    }
#endif
    __syncthreads();
    const int wave = threadIdx.x >> 5;
    const int rowBase = (blockIdx.x * (blockDim.x >> 5) + wave) * 16;
    if (rowBase < Mout) {                                // wave-uniform guard (EXEC all-ones)
        __builtin_prefetch(in + (size_t)(2 * rowBase) * HID, 0, 0);
        block16(in, out, wl, wr, bias, rowBase);
    }
}

// Final 8 levels (128 -> 1 nodes) fused in a single workgroup, h in LDS ping-pong.
__global__ void RecNN_tail_kernel(const _Float16* __restrict__ inG, float* __restrict__ out,
                                  const _Float16* __restrict__ WlT,
                                  const _Float16* __restrict__ WrT,
                                  const float* __restrict__ bias) {
    extern __shared__ char smem[];
    _Float16* ping = (_Float16*)smem;                    // 128 rows * 128 = 32 KB
    _Float16* pong = ping + 128 * HID;                   // 64 rows * 128 = 16 KB
    const int wave = threadIdx.x >> 5;
    const int nw   = blockDim.x >> 5;

    const _Float16* src = inG;                           // 256 rows in global (L2-hot)
    _Float16* dst = ping;
    const _Float16* last = ping;
    int Mout = 128;
#pragma unroll 1
    for (int lvl = 0; lvl < 8; ++lvl) {
        const int nb = (Mout + 15) >> 4;
        for (int b = wave; b < nb; b += nw)              // wave-uniform
            block16(src, dst, WlT, WrT, bias, b * 16);
        __syncthreads();
        last = dst;
        src  = dst;
        dst  = (dst == ping) ? pong : ping;
        Mout >>= 1;
    }
    if (threadIdx.x < HID) out[threadIdx.x] = (float)last[threadIdx.x];
}

// ---------------------------------------------------------------------------
// Launch
// ---------------------------------------------------------------------------
extern "C" void kernel_launch(void* const* d_in, const int* in_sizes, int n_in,
                              void* d_out, int out_size, void* d_ws, size_t ws_size,
                              hipStream_t stream) {
    const float* leaf_x = (const float*)d_in[0];
    const float* W_in   = (const float*)d_in[1];
    const float* b_in   = (const float*)d_in[2];
    const float* W_l    = (const float*)d_in[3];
    const float* b_l    = (const float*)d_in[4];
    const float* W_r    = (const float*)d_in[5];
    const float* b_r    = (const float*)d_in[6];

    char* ws = (char*)d_ws;
    _Float16* WinT  = (_Float16*)(ws);                                 // 16 KB
    _Float16* WlT   = (_Float16*)(ws + 16384);                         // 32 KB
    _Float16* WrT   = (_Float16*)(ws + 16384 + 32768);                 // 32 KB
    float*    biasL = (float*)(ws + 81920);                            // 512 B
    _Float16* hA    = (_Float16*)(ws + 131072);                        // 64 MB
    _Float16* hB    = (_Float16*)(ws + 131072 + (size_t)N_LEAVES * HID * sizeof(_Float16));

    RecNN_prep_kernel<<<(HID * HID + 255) / 256, 256, 0, stream>>>(
        W_in, W_l, W_r, b_l, b_r, WinT, WlT, WrT, biasL);

    RecNN_leaf_kernel<<<N_LEAVES / 128, 256, HID * IN_K * sizeof(_Float16), stream>>>(
        leaf_x, hA, WinT, b_in);

    _Float16* cur = hA;
    _Float16* nxt = hB;
    for (int Mout = N_LEAVES / 2; Mout >= 256; Mout >>= 1) {
        RecNN_level_kernel<<<(Mout + 127) / 128, 256, 2 * HID * HID * sizeof(_Float16), stream>>>(
            cur, nxt, WlT, WrT, biasL, Mout);
        _Float16* t = cur; cur = nxt; nxt = t;
    }

    RecNN_tail_kernel<<<1, 256, (128 + 64) * HID * sizeof(_Float16), stream>>>(
        cur, (float*)d_out, WlT, WrT, biasL);
}
